// RNNModelScratch_12472585028186
// MI455X (gfx1250) — compile-verified, matrix operand order checked by reading, not run on hardware
//
#include <hip/hip_runtime.h>
#include <hip/hip_bf16.h>
#include <math.h>

// Problem sizes (match reference)
#define VOCAB 2048
#define HID   1024
#define BATCH 64
#define STEPS 512

// Persistent-kernel grid split
#define NWG_REC    8                 // 8 WG x 8 waves = 64 waves -> 64 rec strips (4 Mtiles x 16 Nstrips)
#define NWG_PROJ   16                // 16 WG x 8 waves = 128 waves -> 128 proj strips (4 x 32)
#define NWG        (NWG_REC + NWG_PROJ)
#define WG_THREADS 256
#define WAVES_PER_WG 8

typedef __attribute__((ext_vector_type(16))) __bf16 v16bf;
typedef __attribute__((ext_vector_type(8)))  float  v8f;

union BF16Tile {
    uint4 u[2];   // 2 x 16B = 16 bf16
    v16bf v;
};

__device__ __forceinline__ unsigned short f2bf(float f) {
    unsigned u = __float_as_uint(f);
    u += 0x7FFFu + ((u >> 16) & 1u);       // round-to-nearest-even
    return (unsigned short)(u >> 16);
}

// ---------------------------------------------------------------------------
// Prep: convert W_hh, W_hq to bf16, transposed to [N][K] so B-operand tiles
// load as contiguous 16B chunks. Also convert initial H to bf16 ping buffer.
// ---------------------------------------------------------------------------
__global__ void prep_weights(const float* __restrict__ Whh,
                             const float* __restrict__ Whq,
                             const float* __restrict__ h0,
                             unsigned short* __restrict__ whh_t,
                             unsigned short* __restrict__ whq_t,
                             unsigned short* __restrict__ hbuf) {
    const int total = HID * HID + VOCAB * HID + BATCH * HID;
    for (int i = blockIdx.x * blockDim.x + threadIdx.x; i < total;
         i += gridDim.x * blockDim.x) {
        if (i < HID * HID) {
            int n = i / HID, k = i % HID;                  // whh_t[n][k] = Whh[k][n]
            whh_t[i] = f2bf(Whh[(size_t)k * HID + n]);
        } else if (i < HID * HID + VOCAB * HID) {
            int j = i - HID * HID;
            int n = j / HID, k = j % HID;                  // whq_t[n][k] = Whq[k][n]
            whq_t[j] = f2bf(Whq[(size_t)k * VOCAB + n]);
        } else {
            int j = i - HID * HID - VOCAB * HID;           // hbuf[0][b][k] = h0
            hbuf[j] = f2bf(h0[j]);
        }
    }
}

// ---------------------------------------------------------------------------
// One 16x64 output strip: acc[j] (j=0..3) += A(16xK) * B(Kx16), K = HID.
// Abase: &A[m0 + (lane&15)][ (lane>>4)*8 ]   (bf16 row-major, ld = HID)
// Bbase: &Bt[n0 + (lane&15)][ (lane>>4)*8 ]  (bf16 [N][K],   ld = HID)
// ---------------------------------------------------------------------------
__device__ __forceinline__ void gemm_16x64(const unsigned short* __restrict__ Abase,
                                           const unsigned short* __restrict__ Bbase,
                                           v8f acc[4]) {
    for (int k = 0; k < HID; k += 32) {
        BF16Tile A;
        A.u[0] = *(const uint4*)(Abase + k);
        A.u[1] = *(const uint4*)(Abase + k + 16);
#pragma unroll
        for (int j = 0; j < 4; ++j) {
            const unsigned short* bp = Bbase + (size_t)j * 16 * HID;
            BF16Tile B;
            B.u[0] = *(const uint4*)(bp + k);
            B.u[1] = *(const uint4*)(bp + k + 16);
            acc[j] = __builtin_amdgcn_wmma_f32_16x16x32_bf16(
                false, A.v, false, B.v, (short)0, acc[j], false, false);
        }
    }
}

// ---------------------------------------------------------------------------
// Persistent RNN kernel. 24 workgroups, device-scope one-shot barrier per step.
//   rec  WGs (0..7):  H^(t+1) = tanh(gather(Wxh, X[:,t]) + H^(t) Whh + bh)
//   proj WGs (8..23): Y_{t-1} = H^(t) Whq + bq        (lags one step -> overlap)
// ---------------------------------------------------------------------------
__global__ __launch_bounds__(WG_THREADS)
void rnn_persistent(const int* __restrict__ X,
                    const float* __restrict__ Wxh,
                    const float* __restrict__ bh,
                    const float* __restrict__ bq,
                    const unsigned short* __restrict__ whh_t,
                    const unsigned short* __restrict__ whq_t,
                    unsigned short* __restrict__ hbuf,   // [2][BATCH][HID] bf16
                    unsigned* __restrict__ counters,     // [STEPS] one-shot
                    float* __restrict__ out) {           // Y then H_final
    const int wg   = blockIdx.x;
    const int lane = threadIdx.x & 31;
    const int wave = threadIdx.x >> 5;
    const bool isRec = (wg < NWG_REC);
    const int half = lane >> 4;      // 0: K{0..7,16..23}, 1: K{8..15,24..31}
    const int l15  = lane & 15;

    float* outH = out + (size_t)STEPS * BATCH * VOCAB;

    int m0, n0;
    if (isRec) {
        int w = wg * WAVES_PER_WG + wave;              // 0..63
        m0 = (w >> 4) * 16;                            // 4 M-tiles over BATCH
        n0 = (w & 15) * 64;                            // 16 strips over HID
    } else {
        int w = (wg - NWG_REC) * WAVES_PER_WG + wave;  // 0..127
        m0 = (w >> 5) * 16;
        n0 = (w & 31) * 64;                            // 32 strips over VOCAB
    }
    const int rowBase = m0 + 8 * half;                 // C/D layout: M = m0 + 8*half + r

    for (int t = 0; t <= STEPS; ++t) {
        const unsigned short* Hcur = hbuf + (size_t)(t & 1) * (BATCH * HID);
        unsigned short*       Hnxt = hbuf + (size_t)((t + 1) & 1) * (BATCH * HID);
        const unsigned short* Arow = Hcur + (size_t)(m0 + l15) * HID + half * 8;

        if (isRec) {
            if (t < STEPS) {
                v8f acc[4] = {{}, {}, {}, {}};
                gemm_16x64(Arow, whh_t + (size_t)(n0 + l15) * HID + half * 8, acc);

                int tok[8];
#pragma unroll
                for (int r = 0; r < 8; ++r)
                    tok[r] = X[(size_t)(rowBase + r) * STEPS + t];
#pragma unroll
                for (int j = 0; j < 4; ++j) {
                    int n = n0 + j * 16 + l15;
                    float bias = bh[n];
#pragma unroll
                    for (int r = 0; r < 8; ++r) {
                        int row = rowBase + r;
                        float v = acc[j][r] + Wxh[(size_t)tok[r] * HID + n] + bias;
                        float h = tanhf(v);
                        Hnxt[(size_t)row * HID + n] = f2bf(h);
                        if (t == STEPS - 1)
                            outH[(size_t)row * HID + n] = h;   // H_final (f32)
                    }
                }
            }
        } else {
            if (t >= 1) {  // project H^(t) -> Y_{t-1}
                v8f acc[4] = {{}, {}, {}, {}};
                gemm_16x64(Arow, whq_t + (size_t)(n0 + l15) * HID + half * 8, acc);
#pragma unroll
                for (int j = 0; j < 4; ++j) {
                    int n = n0 + j * 16 + l15;
                    float bias = bq[n];
#pragma unroll
                    for (int r = 0; r < 8; ++r) {
                        int b = rowBase + r;
                        out[((size_t)(t - 1) * BATCH + b) * VOCAB + n] = acc[j][r] + bias;
                    }
                }
            }
        }

        if (t < STEPS) {  // device-scope barrier (one-shot counter per step)
            __threadfence();
            __syncthreads();
            if (threadIdx.x == 0) {
                __hip_atomic_fetch_add(&counters[t], 1u, __ATOMIC_ACQ_REL,
                                       __HIP_MEMORY_SCOPE_AGENT);
                while (__hip_atomic_load(&counters[t], __ATOMIC_ACQUIRE,
                                         __HIP_MEMORY_SCOPE_AGENT) < (unsigned)NWG) {
                    __builtin_amdgcn_s_sleep(8);
                }
            }
            __syncthreads();
        }
    }
}

// ---------------------------------------------------------------------------
// Workspace layout (bytes):
//   [0, 4096)                       step barrier counters (memset 0 each call)
//   [4096, +2MB)                    whh_t bf16 [HID][HID]
//   [.., +4MB)                      whq_t bf16 [VOCAB][HID]
//   [.., +256KB)                    hbuf  bf16 [2][BATCH][HID]
// ---------------------------------------------------------------------------
extern "C" void kernel_launch(void* const* d_in, const int* in_sizes, int n_in,
                              void* d_out, int out_size, void* d_ws, size_t ws_size,
                              hipStream_t stream) {
    const int*   X   = (const int*)  d_in[0];
    const float* h0  = (const float*)d_in[1];
    const float* Wxh = (const float*)d_in[2];
    const float* Whh = (const float*)d_in[3];
    const float* bh  = (const float*)d_in[4];
    const float* Whq = (const float*)d_in[5];
    const float* bq  = (const float*)d_in[6];
    float* out = (float*)d_out;

    char* ws = (char*)d_ws;
    unsigned*       counters = (unsigned*)ws;
    unsigned short* whh_t = (unsigned short*)(ws + 4096);
    unsigned short* whq_t = (unsigned short*)(ws + 4096 + (size_t)HID * HID * 2);
    unsigned short* hbuf  = (unsigned short*)(ws + 4096 + (size_t)HID * HID * 2
                                                       + (size_t)VOCAB * HID * 2);

    hipMemsetAsync(counters, 0, 4096, stream);
    prep_weights<<<2048, 256, 0, stream>>>(Whh, Whq, h0, whh_t, whq_t, hbuf);
    rnn_persistent<<<NWG, WG_THREADS, 0, stream>>>(X, Wxh, bh, bq,
                                                   whh_t, whq_t, hbuf, counters, out);
}
